// Qwen3_5Attention_1563368096313
// MI455X (gfx1250) — matrix operand order repaired and checked by reading, not compiled, not run on hardware
//
#include <hip/hip_runtime.h>
#include <hip/hip_bf16.h>
#include <stdint.h>

// ---------------------------------------------------------------------------
// Qwen3.5 attention layer for MI455X (gfx1250, wave32, WMMA).
//   BSZ=4 SEQ=2048 HID=2048 NH=16 NKV=8 HD=128  (n_rep = 2)
// Pipeline: [1] QKV GEMM + bias + per-head RMSNorm -> bf16 Q/K/V^T in d_ws
//           [2] flash attention (online softmax)   -> bf16 attn out in d_ws
//           [3] O-projection GEMM                  -> fp32 d_out
// All matmuls use v_wmma_f32_16x16x32_bf16 with fp32 accumulation.
// bf16 operand staging uses GLOBAL_LOAD_ASYNC_TO_LDS_B128 (ASYNCcnt) when the
// toolchain exposes the builtin, with split waits to overlap copy and math.
// ---------------------------------------------------------------------------

#define BSZ 4
#define SEQ 2048
#define HID 2048
#define NH 16
#define NKV 8
#define HD 128
#define QKV_N ((NH + 2 * NKV) * HD) /* 4096 */
#define RMS_EPS 1e-6f
#define ATTN_SCALE 0.08838834764831845f /* 128^-0.5 */

#if defined(__has_builtin)
#if __has_builtin(__builtin_amdgcn_global_load_async_to_lds_b128)
#define ASYNC_CP 1
#endif
#endif
#ifndef ASYNC_CP
#define ASYNC_CP 0
#endif

typedef __attribute__((ext_vector_type(16))) __bf16 v16bf;
typedef __attribute__((ext_vector_type(8)))  float  v8f;
typedef int v4i_vec __attribute__((vector_size(16)));

union BF16Frag {
  v16bf v;
  uint4 q[2];
};

__device__ __forceinline__ unsigned short f2bf(float f) {
  union { float f; unsigned u; } x;
  x.f = f;
  return (unsigned short)((x.u + 0x7FFFu + ((x.u >> 16) & 1u)) >> 16);
}

__device__ __forceinline__ v8f vzero8() {
  v8f z;
#pragma unroll
  for (int i = 0; i < 8; ++i) z[i] = 0.0f;
  return z;
}

__device__ __forceinline__ v8f wmma_bf16(const BF16Frag& a, const BF16Frag& b, v8f c) {
  // (neg_a, A, neg_b, B, c_mod, C, reuse_a, reuse_b)
  return __builtin_amdgcn_wmma_f32_16x16x32_bf16(false, a.v, false, b.v, (short)0, c,
                                                 false, false);
}

__device__ __forceinline__ void pack4bf(unsigned short* p, float4 f) {
  uint2 t;
  t.x = (unsigned)f2bf(f.x) | ((unsigned)f2bf(f.y) << 16);
  t.y = (unsigned)f2bf(f.z) | ((unsigned)f2bf(f.w) << 16);
  *(uint2*)p = t;
}

// 16-byte global -> LDS copy. Async (no VGPR round trip) when available.
__device__ __forceinline__ void cp_lds16(unsigned short* l, const unsigned short* g) {
#if ASYNC_CP
  __builtin_amdgcn_global_load_async_to_lds_b128(
      (__attribute__((address_space(1))) v4i_vec*)(uintptr_t)g,
      (__attribute__((address_space(3))) v4i_vec*)(unsigned)(uintptr_t)l, 0, 0);
#else
  *(uint4*)l = *(const uint4*)g;
#endif
}

template <int N>
__device__ __forceinline__ void async_wait() {
#if ASYNC_CP
#if defined(__has_builtin) && __has_builtin(__builtin_amdgcn_s_wait_asynccnt)
  __builtin_amdgcn_s_wait_asynccnt((short)N);
#else
  asm volatile("s_wait_asynccnt %0" ::"n"(N) : "memory");
#endif
#endif
}

// ---------------------------------------------------------------------------
// Kernel 1: QKV GEMM (M=8192, N=4096, K=2048) + bias + per-head RMS norm.
// Block tile 128x128 (one head per block in N). 256 threads = 8 waves (4x2).
// Each wave: 32x64 = 2x4 tiles of 16x16. K staged in LDS as bf16, 64 per step.
// ---------------------------------------------------------------------------
__global__ __launch_bounds__(256) void qkv_rms_kernel(
    const float* __restrict__ hidden, const float* __restrict__ qkv_w,
    const float* __restrict__ qkv_b, const float* __restrict__ qnw,
    const float* __restrict__ knw, unsigned short* __restrict__ qws,
    unsigned short* __restrict__ kws, unsigned short* __restrict__ vws) {
  __shared__ unsigned short As[128 * 64];
  __shared__ unsigned short Bs[128 * 64];
  __shared__ float rowsq[128][2];

  const int tid  = threadIdx.x;
  const int wave = tid >> 5;
  const int lane = tid & 31;
  const int l16  = lane & 15;
  const int hi   = lane >> 4;   // lane half: selects K sub-chunk in A/B frags
  const int wm   = wave >> 1;   // 0..3 (M)
  const int wn   = wave & 1;    // 0..1 (N)

  const int m0   = blockIdx.x * 128;
  const int n0   = blockIdx.y * 128;
  const int head = n0 >> 7;     // 0..15 q, 16..23 k, 24..31 v

  v8f acc[2][4];
#pragma unroll
  for (int mt = 0; mt < 2; ++mt)
#pragma unroll
    for (int nt = 0; nt < 4; ++nt) acc[mt][nt] = vzero8();

  for (int k0 = 0; k0 < HID; k0 += 64) {
    // stage A (hidden) 128x64 fp32 -> bf16
#pragma unroll
    for (int i = 0; i < 8; ++i) {
      int c = tid + i * 256;           // 2048 float4 chunks
      int row = c >> 4, part = c & 15;
      const float* src = hidden + (size_t)(m0 + row) * HID + k0 + part * 4;
      float4 f = *(const float4*)src;
      if (k0 + 64 < HID) __builtin_prefetch(src + 64, 0, 0);
      pack4bf(As + row * 64 + part * 4, f);
    }
    // stage B (qkv_w rows = output cols; K contiguous) 128x64 fp32 -> bf16
#pragma unroll
    for (int i = 0; i < 8; ++i) {
      int c = tid + i * 256;
      int row = c >> 4, part = c & 15;
      const float* src = qkv_w + (size_t)(n0 + row) * HID + k0 + part * 4;
      float4 f = *(const float4*)src;
      if (k0 + 64 < HID) __builtin_prefetch(src + 64, 0, 0);
      pack4bf(Bs + row * 64 + part * 4, f);
    }
    __syncthreads();

#pragma unroll
    for (int ks = 0; ks < 2; ++ks) {
      BF16Frag a[2], b[4];
#pragma unroll
      for (int mt = 0; mt < 2; ++mt) {
        int row  = wm * 32 + mt * 16 + l16;
        int base = row * 64 + ks * 32 + hi * 16;
        a[mt].q[0] = *(const uint4*)(As + base);
        a[mt].q[1] = *(const uint4*)(As + base + 8);
      }
#pragma unroll
      for (int nt = 0; nt < 4; ++nt) {
        int col  = wn * 64 + nt * 16 + l16;
        int base = col * 64 + ks * 32 + hi * 16;
        b[nt].q[0] = *(const uint4*)(Bs + base);
        b[nt].q[1] = *(const uint4*)(Bs + base + 8);
      }
#pragma unroll
      for (int mt = 0; mt < 2; ++mt)
#pragma unroll
        for (int nt = 0; nt < 4; ++nt) acc[mt][nt] = wmma_bf16(a[mt], b[nt], acc[mt][nt]);
    }
    __syncthreads();
  }

  // epilogue: bias, per-row sum of squares over this head's 128 cols
  float bias[4], nw[4];
#pragma unroll
  for (int nt = 0; nt < 4; ++nt) {
    int d    = wn * 64 + nt * 16 + l16;
    bias[nt] = qkv_b[n0 + d];
    nw[nt]   = (head < NH) ? qnw[d] : ((head < NH + NKV) ? knw[d] : 1.0f);
  }
#pragma unroll
  for (int mt = 0; mt < 2; ++mt) {
    float ssq[8];
#pragma unroll
    for (int r = 0; r < 8; ++r) {
      float s = 0.0f;
#pragma unroll
      for (int nt = 0; nt < 4; ++nt) {
        acc[mt][nt][r] += bias[nt];
        s += acc[mt][nt][r] * acc[mt][nt][r];
      }
      ssq[r] = s;
    }
#pragma unroll
    for (int m = 1; m < 16; m <<= 1)
#pragma unroll
      for (int r = 0; r < 8; ++r) ssq[r] += __shfl_xor(ssq[r], m, 32);
    if (l16 == 0) {
#pragma unroll
      for (int r = 0; r < 8; ++r) rowsq[wm * 32 + mt * 16 + r + hi * 8][wn] = ssq[r];
    }
  }
  __syncthreads();

#pragma unroll
  for (int mt = 0; mt < 2; ++mt) {
#pragma unroll
    for (int r = 0; r < 8; ++r) {
      int rowl  = wm * 32 + mt * 16 + r + hi * 8;
      float tot = rowsq[rowl][0] + rowsq[rowl][1];
      float rms = rsqrtf(tot * (1.0f / (float)HD) + RMS_EPS);
      int m = m0 + rowl;
      int b = m >> 11;       // /SEQ
      int s = m & (SEQ - 1);
#pragma unroll
      for (int nt = 0; nt < 4; ++nt) {
        int d   = wn * 64 + nt * 16 + l16;
        float v = acc[mt][nt][r];
        if (head < NH + NKV) v *= rms * nw[nt];
        unsigned short hbits = f2bf(v);
        if (head < NH) {
          qws[(((size_t)(b * NH + head)) * SEQ + s) * HD + d] = hbits;
        } else if (head < NH + NKV) {
          kws[(((size_t)(b * NKV + head - NH)) * SEQ + s) * HD + d] = hbits;
        } else {
          // V stored transposed [B,Hkv,D,S] for the PV B-operand
          vws[(((size_t)(b * NKV + head - NH - NKV)) * HD + d) * SEQ + s] = hbits;
        }
      }
    }
  }
}

// ---------------------------------------------------------------------------
// Kernel 2: causal flash attention, GQA n_rep=2, online softmax.
// Grid: (B*NH, SEQ/128). Block 256 thr = 8 waves, wave handles 16 q rows.
// K block + V^T block async-staged in LDS; V copy overlaps the score math
// (wait ASYNCcnt<=8 for K, ASYNCcnt==0 only before PV). P reuses the K buffer.
// ---------------------------------------------------------------------------
__global__ __launch_bounds__(256) void flash_attn_kernel(
    const unsigned short* __restrict__ qws, const unsigned short* __restrict__ kws,
    const unsigned short* __restrict__ vws, unsigned short* __restrict__ ows) {
  __shared__ unsigned short kbuf[128 * 128];  // K block, then P
  __shared__ unsigned short vbuf[128 * 128];  // V^T block

  const int tid  = threadIdx.x;
  const int wave = tid >> 5;
  const int lane = tid & 31;
  const int l16  = lane & 15;
  const int hi   = lane >> 4;

  const int bh  = blockIdx.x;
  const int b   = bh >> 4;
  const int h   = bh & 15;
  const int kvh = h >> 1;           // repeat_interleave(2)
  const int qt  = blockIdx.y;
  const int q0  = qt * 128;

  // Q fragments for this wave's 16 rows (resident across all K blocks)
  const unsigned short* qbase =
      qws + (((size_t)(b * NH + h)) * SEQ + q0 + wave * 16 + l16) * HD;
  BF16Frag qf[4];
#pragma unroll
  for (int ks = 0; ks < 4; ++ks) {
    int base = ks * 32 + hi * 16;
    qf[ks].q[0] = *(const uint4*)(qbase + base);
    qf[ks].q[1] = *(const uint4*)(qbase + base + 8);
  }

  v8f of[8];
#pragma unroll
  for (int nt = 0; nt < 8; ++nt) of[nt] = vzero8();
  float mrun[8], lrun[8];
#pragma unroll
  for (int r = 0; r < 8; ++r) { mrun[r] = -1e30f; lrun[r] = 0.0f; }

  const size_t kgbase = ((size_t)(b * NKV + kvh)) * SEQ * HD;  // + s*HD + d
  const size_t vgbase = ((size_t)(b * NKV + kvh)) * HD * SEQ;  // + d*SEQ + s

  for (int kb = 0; kb <= qt; ++kb) {
    // issue K block copy [128 s][128 d] (8 async ops per thread)
#pragma unroll
    for (int i = 0; i < 8; ++i) {
      int c = tid + i * 256;  // 2048 chunks of 8 halves
      cp_lds16(kbuf + c * 8, kws + kgbase + (size_t)kb * 128 * HD + c * 8);
    }
    // issue V^T block copy [128 d][128 s] (8 more async ops)
#pragma unroll
    for (int i = 0; i < 8; ++i) {
      int c = tid + i * 256;
      int d = c >> 4, part = c & 15;
      cp_lds16(vbuf + d * 128 + part * 8,
               vws + vgbase + (size_t)d * SEQ + kb * 128 + part * 8);
    }
    async_wait<8>();   // K complete (async loads retire in order); V in flight
    __syncthreads();

    // scores: 16 x 128 per wave = 8 n-tiles, K over head dim (4 x 32)
    v8f sf[8];
#pragma unroll
    for (int nt = 0; nt < 8; ++nt) sf[nt] = vzero8();
#pragma unroll
    for (int ks = 0; ks < 4; ++ks) {
#pragma unroll
      for (int nt = 0; nt < 8; ++nt) {
        BF16Frag bb;
        int col  = nt * 16 + l16;
        int base = col * 128 + ks * 32 + hi * 16;
        bb.q[0]  = *(const uint4*)(kbuf + base);
        bb.q[1]  = *(const uint4*)(kbuf + base + 8);
        sf[nt]   = wmma_bf16(qf[ks], bb, sf[nt]);
      }
    }

    // scale + causal mask (only diagonal block needs masking)
    const bool diag = (kb == qt);
#pragma unroll
    for (int nt = 0; nt < 8; ++nt) {
#pragma unroll
      for (int r = 0; r < 8; ++r) {
        float v = sf[nt][r] * ATTN_SCALE;
        int rowl = wave * 16 + r + hi * 8;
        int col  = nt * 16 + l16;
        if (diag && col > rowl) v = -1e30f;
        sf[nt][r] = v;
      }
    }

    // online softmax: row max / exp / row sum (rows live in 16-lane halves)
    float rescale[8];
#pragma unroll
    for (int r = 0; r < 8; ++r) {
      float mx = sf[0][r];
#pragma unroll
      for (int nt = 1; nt < 8; ++nt) mx = fmaxf(mx, sf[nt][r]);
#pragma unroll
      for (int m = 1; m < 16; m <<= 1) mx = fmaxf(mx, __shfl_xor(mx, m, 32));
      float mnew = fmaxf(mrun[r], mx);
      rescale[r] = __expf(mrun[r] - mnew);
      mrun[r]    = mnew;
    }
    float rs[8];
#pragma unroll
    for (int r = 0; r < 8; ++r) rs[r] = 0.0f;
#pragma unroll
    for (int nt = 0; nt < 8; ++nt) {
#pragma unroll
      for (int r = 0; r < 8; ++r) {
        float p = __expf(sf[nt][r] - mrun[r]);
        sf[nt][r] = p;
        rs[r] += p;
      }
    }
#pragma unroll
    for (int m = 1; m < 16; m <<= 1)
#pragma unroll
      for (int r = 0; r < 8; ++r) rs[r] += __shfl_xor(rs[r], m, 32);
#pragma unroll
    for (int r = 0; r < 8; ++r) lrun[r] = lrun[r] * rescale[r] + rs[r];
#pragma unroll
    for (int nt = 0; nt < 8; ++nt)
#pragma unroll
      for (int r = 0; r < 8; ++r) of[nt][r] *= rescale[r];

    async_wait<0>();   // V block complete for this wave
    __syncthreads();   // all waves: V complete + done reading kbuf

    // write P (bf16) into this wave's 16-row slice of kbuf
#pragma unroll
    for (int nt = 0; nt < 8; ++nt) {
#pragma unroll
      for (int r = 0; r < 8; ++r) {
        int rowl = wave * 16 + r + hi * 8;
        int col  = nt * 16 + l16;
        kbuf[rowl * 128 + col] = f2bf(sf[nt][r]);
      }
    }

    // PV: A = P (K over s_k, 4 x 32), B = V^T tiles (8 d-tiles)
#pragma unroll
    for (int ks = 0; ks < 4; ++ks) {
      BF16Frag pa;
      int row  = wave * 16 + l16;
      int base = row * 128 + ks * 32 + hi * 16;
      pa.q[0]  = *(const uint4*)(kbuf + base);
      pa.q[1]  = *(const uint4*)(kbuf + base + 8);
#pragma unroll
      for (int nt = 0; nt < 8; ++nt) {
        BF16Frag bb;
        int col = nt * 16 + l16;
        int vb  = col * 128 + ks * 32 + hi * 16;
        bb.q[0] = *(const uint4*)(vbuf + vb);
        bb.q[1] = *(const uint4*)(vbuf + vb + 8);
        of[nt]  = wmma_bf16(pa, bb, of[nt]);
      }
    }
    __syncthreads();  // protect kbuf/vbuf before next block's staging
  }

  // normalize and write attention output [B,S,NH*HD] bf16
#pragma unroll
  for (int nt = 0; nt < 8; ++nt) {
#pragma unroll
    for (int r = 0; r < 8; ++r) {
      int rowl = wave * 16 + r + hi * 8;
      int s    = q0 + rowl;
      int d    = nt * 16 + l16;
      float v  = of[nt][r] * (1.0f / lrun[r]);
      ows[((size_t)(b * SEQ + s)) * (NH * HD) + h * HD + d] = f2bf(v);
    }
  }
}

// ---------------------------------------------------------------------------
// Kernel 3: O projection (M=8192, N=2048, K=2048), A bf16 (async-staged),
// B fp32->bf16, fp32 output directly to d_out.
// ---------------------------------------------------------------------------
__global__ __launch_bounds__(256) void oproj_kernel(
    const unsigned short* __restrict__ attn, const float* __restrict__ o_w,
    float* __restrict__ out) {
  __shared__ unsigned short As[128 * 64];
  __shared__ unsigned short Bs[128 * 64];

  const int tid  = threadIdx.x;
  const int wave = tid >> 5;
  const int lane = tid & 31;
  const int l16  = lane & 15;
  const int hi   = lane >> 4;
  const int wm   = wave >> 1;
  const int wn   = wave & 1;

  const int m0 = blockIdx.x * 128;
  const int n0 = blockIdx.y * 128;

  v8f acc[2][4];
#pragma unroll
  for (int mt = 0; mt < 2; ++mt)
#pragma unroll
    for (int nt = 0; nt < 4; ++nt) acc[mt][nt] = vzero8();

  for (int k0 = 0; k0 < NH * HD; k0 += 64) {
    // stage A (already bf16) via async copy: 128x64 halves = 1024 16B chunks
#pragma unroll
    for (int i = 0; i < 4; ++i) {
      int c = tid + i * 256;
      int row = c >> 3, part = c & 7;
      cp_lds16(As + row * 64 + part * 8,
               attn + (size_t)(m0 + row) * (NH * HD) + k0 + part * 8);
    }
    // stage B (o_w rows = output cols, K contiguous) fp32 -> bf16,
    // overlapping with the async A copy above.
#pragma unroll
    for (int i = 0; i < 8; ++i) {
      int c = tid + i * 256;
      int row = c >> 4, part = c & 15;
      const float* src = o_w + (size_t)(n0 + row) * (NH * HD) + k0 + part * 4;
      float4 f = *(const float4*)src;
      if (k0 + 64 < NH * HD) __builtin_prefetch(src + 64, 0, 0);
      pack4bf(Bs + row * 64 + part * 4, f);
    }
    async_wait<0>();
    __syncthreads();

#pragma unroll
    for (int ks = 0; ks < 2; ++ks) {
      BF16Frag a[2], b[4];
#pragma unroll
      for (int mt = 0; mt < 2; ++mt) {
        int row  = wm * 32 + mt * 16 + l16;
        int base = row * 64 + ks * 32 + hi * 16;
        a[mt].q[0] = *(const uint4*)(As + base);
        a[mt].q[1] = *(const uint4*)(As + base + 8);
      }
#pragma unroll
      for (int nt = 0; nt < 4; ++nt) {
        int col  = wn * 64 + nt * 16 + l16;
        int base = col * 64 + ks * 32 + hi * 16;
        b[nt].q[0] = *(const uint4*)(Bs + base);
        b[nt].q[1] = *(const uint4*)(Bs + base + 8);
      }
#pragma unroll
      for (int mt = 0; mt < 2; ++mt)
#pragma unroll
        for (int nt = 0; nt < 4; ++nt) acc[mt][nt] = wmma_bf16(a[mt], b[nt], acc[mt][nt]);
    }
    __syncthreads();
  }

#pragma unroll
  for (int mt = 0; mt < 2; ++mt) {
#pragma unroll
    for (int r = 0; r < 8; ++r) {
      int m = m0 + wm * 32 + mt * 16 + r + hi * 8;
#pragma unroll
      for (int nt = 0; nt < 4; ++nt) {
        int col = n0 + wn * 64 + nt * 16 + l16;
        out[(size_t)m * HID + col] = acc[mt][nt][r];
      }
    }
  }
}

// ---------------------------------------------------------------------------
extern "C" void kernel_launch(void* const* d_in, const int* in_sizes, int n_in,
                              void* d_out, int out_size, void* d_ws, size_t ws_size,
                              hipStream_t stream) {
  const float* hidden = (const float*)d_in[0];
  const float* qkv_w  = (const float*)d_in[7];
  const float* qkv_b  = (const float*)d_in[8];
  const float* o_w    = (const float*)d_in[9];
  const float* qnw    = (const float*)d_in[10];
  const float* knw    = (const float*)d_in[11];
  float* out          = (float*)d_out;

  // workspace carve-up (bf16 elements)
  unsigned short* qws = (unsigned short*)d_ws;                       // 32 MB
  unsigned short* kws = qws + (size_t)BSZ * NH * SEQ * HD;           // 16 MB
  unsigned short* vws = kws + (size_t)BSZ * NKV * SEQ * HD;          // 16 MB
  unsigned short* ows = vws + (size_t)BSZ * NKV * SEQ * HD;          // 32 MB

  dim3 blk(256);
  qkv_rms_kernel<<<dim3((BSZ * SEQ) / 128, QKV_N / 128), blk, 0, stream>>>(
      hidden, qkv_w, qkv_b, qnw, knw, qws, kws, vws);
  flash_attn_kernel<<<dim3(BSZ * NH, SEQ / 128), blk, 0, stream>>>(qws, kws, vws, ows);
  oproj_kernel<<<dim3((BSZ * SEQ) / 128, HID / 128), blk, 0, stream>>>(ows, o_w, out);
}
